// TemporalAttention_30185030156476
// MI455X (gfx1250) — matrix operand order repaired and checked by reading, not compiled, not run on hardware
//
#include <hip/hip_runtime.h>
#include <cmath>

typedef __attribute__((ext_vector_type(16))) __bf16 v16bf;
typedef __attribute__((ext_vector_type(8)))  float  v8f;

#define HDIM 512
#define SDIM 2048
#define BDIM 64
#define ROWS 32     // s-rows per block in scores kernel (2 M-tiles of 16)

// ---------------------------------------------------------------------------
// Pack w_enc (attn_w[:, :H], used as w_enc[k,h]) into WMMA B-fragment layout.
// wpk[ct][kk][lane][e] : ct = N-tile (0..31), kk = K-chunk of 32 (0..15),
// B element (k_red, n): k_red = kk*32 + (lane<16 ? e : 16+e); n = lane&15.
// B[k_red][n] = attn_w[(ct*16+n) * 2H + k_red]   (reduction dim = attn_w col)
// ---------------------------------------------------------------------------
__global__ __launch_bounds__(256)
void pack_wenc(const float* __restrict__ attn_w, __bf16* __restrict__ wpk) {
    int o    = blockIdx.x * 256 + threadIdx.x;   // 262144 total
    int e    = o & 15;
    int lane = (o >> 4) & 31;
    int kk   = (o >> 9) & 15;
    int ct   = o >> 13;
    int n    = lane & 15;
    int k    = (lane < 16) ? e : (16 + e);
    int col  = ct * 16 + n;
    int h    = kk * 32 + k;
    wpk[o] = (__bf16)attn_w[(size_t)col * (2 * HDIM) + h];
}

// ---------------------------------------------------------------------------
// t[b,k] = sum_h fin[b,h] * attn_w[k, H+h] + attn_b[k]   (tiny GEMV, fp32)
// ---------------------------------------------------------------------------
__global__ __launch_bounds__(256)
void fin_bias(const float* __restrict__ fin, const float* __restrict__ attn_w,
              const float* __restrict__ attn_b, float* __restrict__ tb) {
    int o = blockIdx.x * 256 + threadIdx.x;      // 64*512
    int k = o & (HDIM - 1);
    int b = o >> 9;
    const float* wf = attn_w + (size_t)k * (2 * HDIM) + HDIM;
    const float* f  = fin + (size_t)b * HDIM;
    float acc = 0.f;
    #pragma unroll 8
    for (int h = 0; h < HDIM; ++h) acc += f[h] * wf[h];
    tb[o] = acc + attn_b[k];
}

// ---------------------------------------------------------------------------
// scores[b,s] = sum_k tanh( (enc[b,s,:] . w_enc[k,:]) + t[b,k] ) * v[k]
// Block: 256 threads (8 waves), 32 s-rows of one batch.
//   LDS A: enc tile as bf16 WMMA A-fragments   [2 mt][16 kk][32 lane][16 e]
//   LDS B: per-kk staged B fragments           [32 ct][32 lane][16 e]
//   (B buffer re-used as per-row score accumulator after the K loop)
// Wave w: mt = w>>2 (16 rows), ct0 = (w&3)*8 (8 column tiles) -> 8 v8f accs.
// ---------------------------------------------------------------------------
__global__ __launch_bounds__(256)
void scores_kernel(const float* __restrict__ enc, const __bf16* __restrict__ wpk,
                   const float* __restrict__ tb, const float* __restrict__ vw,
                   float* __restrict__ scores) {
    __shared__ __bf16 lds_a[ROWS * HDIM];     // 32 KB
    __shared__ __bf16 lds_b[32 * 32 * 16];    // 32 KB, aliased below
    float* lds_s = (float*)lds_b;

    const int tid  = threadIdx.x;
    const int lane = tid & 31;
    const int wave = tid >> 5;
    const int blk  = blockIdx.x;              // 4096 = 64 b * 64 chunks
    const int b    = blk >> 6;
    const int s0   = (blk & 63) * ROWS;

    // ---- Stage enc tile (32 x 512 fp32) -> bf16 A fragments in LDS ----
    const float* encp = enc + ((size_t)b * SDIM + s0) * HDIM;
    #pragma unroll 4
    for (int i = 0; i < 64; ++i) {
        int idx = tid + 256 * i;              // 0..16383, coalesced
        int row = idx >> 9;
        int h   = idx & (HDIM - 1);
        float v = encp[idx];
        int kk = h >> 5, r = h & 31;
        int l, e;
        if (r < 16) { l = 0;  e = r;      }
        else        { l = 16; e = r - 16; }
        int mt = row >> 4, lrow = row & 15;
        lds_a[((mt * 16 + kk) * 32 + (l + lrow)) * 16 + e] = (__bf16)v;
    }
    __syncthreads();

    const int mt  = wave >> 2;                // 0..1
    const int ct0 = (wave & 3) * 8;           // 8 column tiles per wave

    v8f acc[8];
    #pragma unroll
    for (int j = 0; j < 8; ++j)
        #pragma unroll
        for (int r = 0; r < 8; ++r) acc[j][r] = 0.f;

    for (int kk = 0; kk < 16; ++kk) {
        // Stage B(all 32 ct, this kk): 1024 units of 32B, 4 units/thread
        #pragma unroll
        for (int u = 0; u < 4; ++u) {
            int unit = tid + 256 * u;         // 0..1023
            int ct = unit >> 5, ul = unit & 31;
            const uint4* src = (const uint4*)(wpk + (((size_t)ct * 16 + kk) * 32 + ul) * 16);
            uint4* dst = (uint4*)(lds_b + ((ct * 32 + ul) * 16));
            dst[0] = src[0];
            dst[1] = src[1];
        }
        __syncthreads();

        v16bf afrag = *(const v16bf*)(lds_a + ((mt * 16 + kk) * 32 + lane) * 16);
        #pragma unroll
        for (int j = 0; j < 8; ++j) {
            v16bf bfrag = *(const v16bf*)(lds_b + ((ct0 + j) * 32 + lane) * 16);
            acc[j] = __builtin_amdgcn_wmma_f32_16x16x32_bf16(
                false, afrag, false, bfrag, (short)0, acc[j], false, false);
        }
        __syncthreads();
    }

    // ---- bias + tanh + dot with v, per-row partials ----
    // C layout: element r of acc -> row M = r + (lane>=16 ? 8 : 0), col n = lane&15
    const int n  = lane & 15;
    const int hi = (lane >> 4) & 1;
    float part[8];
    #pragma unroll
    for (int r = 0; r < 8; ++r) part[r] = 0.f;
    #pragma unroll
    for (int j = 0; j < 8; ++j) {
        int col    = (ct0 + j) * 16 + n;
        float bias = tb[(size_t)b * HDIM + col];
        float vv   = vw[col];
        #pragma unroll
        for (int r = 0; r < 8; ++r)
            part[r] += tanhf(acc[j][r] + bias) * vv;
    }
    // reduce over the 16 columns (lanes sharing the same row half)
    #pragma unroll
    for (int m = 1; m <= 8; m <<= 1) {
        #pragma unroll
        for (int r = 0; r < 8; ++r) part[r] += __shfl_xor(part[r], m, 32);
    }

    if (tid < ROWS) lds_s[tid] = 0.f;
    __syncthreads();
    if (n == 0) {                             // lanes 0 and 16 of each wave
        #pragma unroll
        for (int r = 0; r < 8; ++r)
            atomicAdd(&lds_s[mt * 16 + hi * 8 + r], part[r]);   // ds_add_f32
    }
    __syncthreads();
    if (tid < ROWS)
        scores[(size_t)b * SDIM + s0 + tid] = lds_s[tid];
}

// ---------------------------------------------------------------------------
// Row softmax over S=2048: one block per batch.
// ---------------------------------------------------------------------------
__global__ __launch_bounds__(256)
void softmax_kernel(const float* __restrict__ scores, float* __restrict__ weights) {
    __shared__ float red[8];
    const int b = blockIdx.x, tid = threadIdx.x;
    const float* sr = scores + (size_t)b * SDIM;
    float v[8];
    float lm = -1e30f;
    #pragma unroll
    for (int i = 0; i < 8; ++i) { v[i] = sr[tid + 256 * i]; lm = fmaxf(lm, v[i]); }
    #pragma unroll
    for (int m = 1; m < 32; m <<= 1) lm = fmaxf(lm, __shfl_xor(lm, m, 32));
    if ((tid & 31) == 0) red[tid >> 5] = lm;
    __syncthreads();
    float bm = red[0];
    #pragma unroll
    for (int i = 1; i < 8; ++i) bm = fmaxf(bm, red[i]);
    float ls = 0.f;
    #pragma unroll
    for (int i = 0; i < 8; ++i) { v[i] = __expf(v[i] - bm); ls += v[i]; }
    #pragma unroll
    for (int m = 1; m < 32; m <<= 1) ls += __shfl_xor(ls, m, 32);
    __syncthreads();
    if ((tid & 31) == 0) red[tid >> 5] = ls;
    __syncthreads();
    float tot = 0.f;
    #pragma unroll
    for (int i = 0; i < 8; ++i) tot += red[i];
    float inv = 1.f / tot;
    #pragma unroll
    for (int i = 0; i < 8; ++i)
        weights[(size_t)b * SDIM + tid + 256 * i] = v[i] * inv;
}

__global__ __launch_bounds__(256)
void zero_out(float* __restrict__ out) {
    out[blockIdx.x * 256 + threadIdx.x] = 0.f;
}

// ---------------------------------------------------------------------------
// context[b,h] = sum_s weights[b,s] * enc[b,s,h]
// 512 blocks = 64 b x 8 s-chunks of 256; partial sums via global f32 atomics.
// ---------------------------------------------------------------------------
#define SCHUNK 256
__global__ __launch_bounds__(256)
void context_kernel(const float* __restrict__ enc, const float* __restrict__ weights,
                    float* __restrict__ out) {
    __shared__ float w[SCHUNK];
    const int b   = blockIdx.x >> 3;
    const int c   = blockIdx.x & 7;
    const int tid = threadIdx.x;
    const int s0  = c * SCHUNK;
    w[tid] = weights[(size_t)b * SDIM + s0 + tid];
    __syncthreads();
    const int h = tid * 2;
    float ax = 0.f, ay = 0.f;
    const float* ep = enc + ((size_t)b * SDIM + s0) * HDIM + h;
    #pragma unroll 8
    for (int s = 0; s < SCHUNK; ++s) {
        float2 e2 = *(const float2*)(ep + (size_t)s * HDIM);
        ax += w[s] * e2.x;
        ay += w[s] * e2.y;
    }
    atomicAdd(&out[b * HDIM + h],     ax);
    atomicAdd(&out[b * HDIM + h + 1], ay);
}

// ---------------------------------------------------------------------------
extern "C" void kernel_launch(void* const* d_in, const int* in_sizes, int n_in,
                              void* d_out, int out_size, void* d_ws, size_t ws_size,
                              hipStream_t stream) {
    const float* enc    = (const float*)d_in[0];   // (64, 2048, 512)
    const float* fin    = (const float*)d_in[1];   // (64, 512)
    const float* attn_w = (const float*)d_in[2];   // (512, 1024)
    const float* attn_b = (const float*)d_in[3];   // (512,)
    const float* v_w    = (const float*)d_in[4];   // (1, 512)
    float* out = (float*)d_out;                    // (64, 512)

    char* ws = (char*)d_ws;
    __bf16* wpk     = (__bf16*)ws;                                   // 512 KB
    float*  tb      = (float*)(ws + 524288);                         // 128 KB
    float*  scores  = (float*)(ws + 524288 + 131072);                // 512 KB
    float*  weights = (float*)(ws + 524288 + 131072 + 524288);       // 512 KB

    pack_wenc   <<<1024, 256, 0, stream>>>(attn_w, wpk);
    fin_bias    <<< 128, 256, 0, stream>>>(fin, attn_w, attn_b, tb);
    scores_kernel<<<4096, 256, 0, stream>>>(enc, wpk, tb, v_w, scores);
    softmax_kernel<<<BDIM, 256, 0, stream>>>(scores, weights);
    zero_out    <<< 128, 256, 0, stream>>>(out);
    context_kernel<<<BDIM * 8, 256, 0, stream>>>(enc, weights, out);
}
